// DeepAngAEVComputer_18124761989631
// MI455X (gfx1250) — matrix operand order repaired and checked by reading, not compiled, not run on hardware
//
#include <hip/hip_runtime.h>
#include <hip/hip_bf16.h>

typedef _Float16 v16h __attribute__((ext_vector_type(16)));
typedef float    v8f  __attribute__((ext_vector_type(8)));
typedef unsigned int u32x4 __attribute__((ext_vector_type(4)));

#define CUTOFF_F 3.5f
#define EPS_F    1e-7f
#define CLIP_MIN 1e-10f

// Transposed f16 weight offsets in d_ws (in halves): WT_l[c * Kp + k]
// l0: 64x32=2048  l1..l4: 64x64=4096  l5: 128x64=8192  l6: 256x128=32768
#define OFF0 0
#define OFF1 2048
#define OFF2 6144
#define OFF3 10240
#define OFF4 14336
#define OFF5 18432
#define OFF6 26624
// total 59392 halves = 118784 bytes in d_ws

// ---------------- dynamic LDS layout (bytes) ----------------
#define SH_PAIRS 0        // 496 * u32              = 1984
#define SH_DMAT  1984     // 1024 * f32             = 4096
#define SH_SROW  6080     // 32 * f32               = 128
#define SH_WALL  6208     // 496 * f32              = 1984
#define SH_ACC   8192     // 256 * f32              = 1024
#define SH_RED   9216     // 4 * f32 (+pad)         = 64
#define SH_FEAT  9280     // 496*32 halves          = 31744
#define SH_WAVE  41024    // 4 waves * 8192 B       = 32768
#define SH_BYTES 73792

__device__ __forceinline__ unsigned int packh(float a, float b) {
  unsigned short ua = __builtin_bit_cast(unsigned short, (_Float16)a);
  unsigned short ub = __builtin_bit_cast(unsigned short, (_Float16)b);
  return (unsigned int)ua | ((unsigned int)ub << 16);
}

// Branch-free tanh on the native trans pipe:
// tanh(x) = sign(x) * (1 - 2/(exp(2|x|)+1)).  exp->inf saturates cleanly to +-1.
__device__ __forceinline__ float fast_tanh(float x) {
  const float ax = fabsf(x);
  const float e  = __expf(2.0f * ax);                       // v_exp_f32 path
  const float y  = 1.0f - 2.0f * __builtin_amdgcn_rcpf(e + 1.0f);
  return copysignf(y, x);
}

// A fragment: 16x32 f16, row-major LDS buffer with row stride K halves.
// ISA layout: lane<16 -> row=lane, halves = K[k0+0..7], K[k0+16..23]
//             lane>=16 -> row=lane-16, halves = K[k0+8..15], K[k0+24..31]
template <int K>
__device__ __forceinline__ v16h load_a(const _Float16* aBuf, int lane, int c) {
  const int row = lane & 15;
  const int sel = ((lane >> 4) & 1) * 8;
  const _Float16* pr = aBuf + row * K + c * 32 + sel;
  v16h av;
  reinterpret_cast<u32x4*>(&av)[0] = *reinterpret_cast<const u32x4*>(pr);
  reinterpret_cast<u32x4*>(&av)[1] = *reinterpret_cast<const u32x4*>(pr + 16);
  return av;
}

// B fragment: 32x16 f16 from transposed weights WT[col * K + k] (global).
// ISA layout: lane = column (mod 16); lanes 0-15 hold K[k0+0..7], K[k0+16..23],
//             lanes 16-31 hold K[k0+8..15], K[k0+24..31].
template <int K>
__device__ __forceinline__ v16h load_b(const _Float16* wt, int lane, int n, int c) {
  const int col = n * 16 + (lane & 15);
  const int sel = ((lane >> 4) & 1) * 8;
  const _Float16* pc = wt + col * K + c * 32 + sel;
  v16h bv;
  reinterpret_cast<u32x4*>(&bv)[0] = *reinterpret_cast<const u32x4*>(pc);
  reinterpret_cast<u32x4*>(&bv)[1] = *reinterpret_cast<const u32x4*>(pc + 16);
  return bv;
}

// One MLP layer on a 16-row tile: D = A(16xK) * WT^T(KxN) ; epilogue per 16-col tile.
// C/D layout: vgpr r, lane L -> row = r + 8*(L>=16), col = L&15.
// NOTE: the n-loop is deliberately NOT unrolled so weight-fragment loads stay
// loop-variant -> LICM cannot hoist ~900 VGPRs of weights across the tile loop
// (which caused scratch spills in the previous round).
template <int K, int N, typename Epi>
__device__ __forceinline__ void run_layer(const _Float16* aBuf, const _Float16* wt,
                                          const float* __restrict__ bias, int lane,
                                          Epi&& epi) {
  constexpr int KC = K / 32;
  v16h a[KC];
#pragma unroll
  for (int c = 0; c < KC; ++c) a[c] = load_a<K>(aBuf, lane, c);
#pragma unroll 1
  for (int n = 0; n < N / 16; ++n) {
    v8f acc = {};
#pragma unroll
    for (int c = 0; c < KC; ++c) {
      v16h bv = load_b<K>(wt, lane, n, c);
      acc = __builtin_amdgcn_wmma_f32_16x16x32_f16(false, a[c], false, bv,
                                                   (short)0, acc, false, false);
    }
    const float bb = bias[n * 16 + (lane & 15)];
    epi(n, acc, bb);
  }
}

// ---------------- weight prep: f32 (KxF) -> f16 transposed (FxKp) ----------------
__global__ void prep_weights(const float* __restrict__ W0, const float* __restrict__ W1,
                             const float* __restrict__ W2, const float* __restrict__ W3,
                             const float* __restrict__ W4, const float* __restrict__ W5,
                             const float* __restrict__ W6, _Float16* __restrict__ ws) {
  const float* W; int K, F, Kp, off;
  switch (blockIdx.y) {
    case 0: W = W0; K = 9;   F = 64;  Kp = 32;  off = OFF0; break;
    case 1: W = W1; K = 64;  F = 64;  Kp = 64;  off = OFF1; break;
    case 2: W = W2; K = 64;  F = 64;  Kp = 64;  off = OFF2; break;
    case 3: W = W3; K = 64;  F = 64;  Kp = 64;  off = OFF3; break;
    case 4: W = W4; K = 64;  F = 64;  Kp = 64;  off = OFF4; break;
    case 5: W = W5; K = 64;  F = 128; Kp = 64;  off = OFF5; break;
    default:W = W6; K = 128; F = 256; Kp = 128; off = OFF6; break;
  }
  const int total = F * Kp;
  for (int idx = blockIdx.x * blockDim.x + threadIdx.x; idx < total;
       idx += gridDim.x * blockDim.x) {
    const int c = idx / Kp;
    const int k = idx - c * Kp;
    const float v = (k < K) ? W[k * F + c] : 0.0f;
    ws[off + idx] = (_Float16)v;
  }
}

// ---------------- main kernel: one workgroup per (b,i) ----------------
__global__ void __launch_bounds__(128)
aev_kernel(const float* __restrict__ Dg, const float* __restrict__ Sg,
           const _Float16* __restrict__ ws,
           const float* __restrict__ b0, const float* __restrict__ b1,
           const float* __restrict__ b2, const float* __restrict__ b3,
           const float* __restrict__ b4, const float* __restrict__ b5,
           const float* __restrict__ b6, float* __restrict__ outG) {
  extern __shared__ __align__(16) char smem[];
  unsigned int* pairs = (unsigned int*)(smem + SH_PAIRS);
  float* Dm          = (float*)(smem + SH_DMAT);
  float* Sr          = (float*)(smem + SH_SROW);
  float* wAll        = (float*)(smem + SH_WALL);
  float* accF        = (float*)(smem + SH_ACC);
  float* red         = (float*)(smem + SH_RED);
  _Float16* featAll  = (_Float16*)(smem + SH_FEAT);
  _Float16* waveMem  = (_Float16*)(smem + SH_WAVE);

  const int tid = threadIdx.x;
  const int lane = tid & 31;
  const int wv = tid >> 5;
  const int bx = blockIdx.x;
  const int b = bx >> 5;
  const int i = bx & 31;

  // ---- stage D, S; zero accumulator; build triplet pair table ----
  for (int q = tid; q < 1024; q += 128) Dm[q] = Dg[b * 1024 + q];
  if (tid < 32) Sr[tid] = Sg[b * 32 + tid];
  for (int q = tid; q < 256; q += 128) accF[q] = 0.0f;
  for (int p = tid; p < 496; p += 128) {
    int pp = p, j = 0;
    while (pp >= 31 - j) { pp -= 31 - j; ++j; }
    pairs[p] = (unsigned)j | ((unsigned)(j + 1 + pp) << 8);
  }
  __syncthreads();

  // ---- features for all 496 triplets of this (b,i) ----
  for (int p = tid; p < 496; p += 128) {
    const unsigned pr = pairs[p];
    const int j = pr & 255, k = (pr >> 8) & 255;
    const float Rij = Dm[i * 32 + j];
    const float Rik = Dm[i * 32 + k];
    const float Rjk = Dm[j * 32 + k];
    const float zi = Sr[i], zj = Sr[j], zk = Sr[k];
    const float rij2 = Rij * Rij, rik2 = Rik * Rik, rjk2 = Rjk * Rjk;
    const float ci = (rij2 + rik2 - rjk2) / fmaxf(2.0f * Rij * Rik, CLIP_MIN);
    const float cj = (rij2 + rjk2 - rik2) / fmaxf(2.0f * Rij * Rjk, CLIP_MIN);
    const float ck = (rik2 + rjk2 - rij2) / fmaxf(2.0f * Rik * Rjk, CLIP_MIN);
    float g0 = Rij + Rik + Rjk;
    float g1 = Rij * Rik + Rij * Rjk + Rik * Rjk;
    float g2 = Rij * Rik * Rjk;
    const float gn = 1.0f / (sqrtf(g0 * g0 + g1 * g1 + g2 * g2) + EPS_F);
    g0 *= gn; g1 *= gn; g2 *= gn;
    float h0 = zi + zj + zk;
    float h1 = ci + cj + ck;
    float h2 = zi * (zj + zk) + zj * zk - ci * (cj + ck) - cj * ck;
    float h3 = zi * (cj + ck) + ci * (zj + zk) + zj * ck + cj * zk;
    float h4 = zi * (zj * zk - cj * ck) - ci * (zj * ck + cj * zk);
    float h5 = zi * (zj * ck + cj * zk) + ci * (zj * zk - cj * ck);
    const float hn = 1.0f / (sqrtf(h0 * h0 + h1 * h1 + h2 * h2 + h3 * h3 +
                                   h4 * h4 + h5 * h5) + EPS_F);
    h0 *= hn; h1 *= hn; h2 *= hn; h3 *= hn; h4 *= hn; h5 *= hn;

    const bool m = (Rij < CUTOFF_F) && (Rij != 0.0f) &&
                   (Rik < CUTOFF_F) && (Rik != 0.0f);
    const float pic = 3.14159265358979f / CUTOFF_F;
    const float w = m ? (0.5f * cosf(pic * Rij) + 0.5f) *
                        (0.5f * cosf(pic * Rik) + 0.5f)
                      : 0.0f;
    wAll[p] = w;

    _Float16* fr = featAll + p * 32;  // K padded to 32
    u32x4 q0 = {packh(g0, g1), packh(g2, h0), packh(h1, h2), packh(h3, h4)};
    u32x4 q1 = {packh(h5, 0.0f), 0u, 0u, 0u};
    u32x4 qz = {0u, 0u, 0u, 0u};
    *reinterpret_cast<u32x4*>(fr)      = q0;
    *reinterpret_cast<u32x4*>(fr + 8)  = q1;
    *reinterpret_cast<u32x4*>(fr + 16) = qz;
    *reinterpret_cast<u32x4*>(fr + 24) = qz;
  }
  __syncthreads();

  // per-wave activation buffers (halves)
  _Float16* A64a = waveMem + wv * 4096;  // 16x64
  _Float16* A64b = A64a + 1024;          // 16x64
  _Float16* A128 = A64a + 2048;          // 16x128

  const _Float16* wt0 = ws + OFF0;
  const _Float16* wt1 = ws + OFF1;
  const _Float16* wt2 = ws + OFF2;
  const _Float16* wt3 = ws + OFF3;
  const _Float16* wt4 = ws + OFF4;
  const _Float16* wt5 = ws + OFF5;
  const _Float16* wt6 = ws + OFF6;

  const int colLo = lane & 15;
  const int rb = (lane >> 4) * 8;

  // 496 = 31 tiles of 16 rows; waves split tiles (wave-uniform loop -> EXEC all-1)
  for (int t = wv; t < 31; t += 4) {
    const int rowBase = t * 16;
    const _Float16* fRow = featAll + rowBase * 32;
    float wr[8];
#pragma unroll
    for (int r = 0; r < 8; ++r) wr[r] = wAll[rowBase + rb + r];

    // L0: feat(9->pad32) -> x_res (A64a)
    run_layer<32, 64>(fRow, wt0, b0, lane, [&](int n, v8f acc, float bb) {
      const int col = n * 16 + colLo;
#pragma unroll
      for (int r = 0; r < 8; ++r)
        A64a[(rb + r) * 64 + col] = (_Float16)fast_tanh(acc[r] + bb);
    });
    // L1: x1 = tanh(x_res@W1+b1); xb1 = x1 + x_res  (A64a -> A64b)
    run_layer<64, 64>(A64a, wt1, b1, lane, [&](int n, v8f acc, float bb) {
      const int col = n * 16 + colLo;
#pragma unroll
      for (int r = 0; r < 8; ++r) {
        const float x1 = fast_tanh(acc[r] + bb);
        A64b[(rb + r) * 64 + col] =
            (_Float16)(x1 + (float)A64a[(rb + r) * 64 + col]);
      }
    });
    // L2: x2 = tanh(xb1@W2+b2)  (A64b -> A64a)
    run_layer<64, 64>(A64b, wt2, b2, lane, [&](int n, v8f acc, float bb) {
      const int col = n * 16 + colLo;
#pragma unroll
      for (int r = 0; r < 8; ++r)
        A64a[(rb + r) * 64 + col] = (_Float16)fast_tanh(acc[r] + bb);
    });
    // L3: x3 = tanh(x2@W3+b3)  (A64a -> A64a; A frags preloaded before stores)
    run_layer<64, 64>(A64a, wt3, b3, lane, [&](int n, v8f acc, float bb) {
      const int col = n * 16 + colLo;
#pragma unroll
      for (int r = 0; r < 8; ++r)
        A64a[(rb + r) * 64 + col] = (_Float16)fast_tanh(acc[r] + bb);
    });
    // L4: x4 = tanh(x3@W4+b4); xb2 = x4 + xb1  (A64a -> A64a, reads A64b)
    run_layer<64, 64>(A64a, wt4, b4, lane, [&](int n, v8f acc, float bb) {
      const int col = n * 16 + colLo;
#pragma unroll
      for (int r = 0; r < 8; ++r) {
        const float x4 = fast_tanh(acc[r] + bb);
        A64a[(rb + r) * 64 + col] =
            (_Float16)(x4 + (float)A64b[(rb + r) * 64 + col]);
      }
    });
    // L5: xb3 = tanh(xb2@W5+b5)  (A64a -> A128)
    run_layer<64, 128>(A64a, wt5, b5, lane, [&](int n, v8f acc, float bb) {
      const int col = n * 16 + colLo;
#pragma unroll
      for (int r = 0; r < 8; ++r)
        A128[(rb + r) * 128 + col] = (_Float16)fast_tanh(acc[r] + bb);
    });
    // L6: out = tanh(xb3@W6+b6); weighted row-sum into shared accumulator
    run_layer<128, 256>(A128, wt6, b6, lane, [&](int n, v8f acc, float bb) {
      const int col = n * 16 + colLo;
      float part = 0.0f;
#pragma unroll
      for (int r = 0; r < 8; ++r) part += fast_tanh(acc[r] + bb) * wr[r];
      atomicAdd(&accF[col], part);  // ds_add_f32
    });
  }
  __syncthreads();

  // ---- L2-normalize GA and write out[b][i][0..255] ----
  float ss = 0.0f;
  for (int c = tid; c < 256; c += 128) { const float v = accF[c]; ss += v * v; }
#pragma unroll
  for (int o = 16; o > 0; o >>= 1) ss += __shfl_xor(ss, o, 32);
  if (lane == 0) red[wv] = ss;
  __syncthreads();
  const float inv = 1.0f / (sqrtf(red[0] + red[1] + red[2] + red[3]) + EPS_F);
  for (int c = tid; c < 256; c += 128) outG[bx * 256 + c] = accF[c] * inv;
}

extern "C" void kernel_launch(void* const* d_in, const int* in_sizes, int n_in,
                              void* d_out, int out_size, void* d_ws, size_t ws_size,
                              hipStream_t stream) {
  (void)in_sizes; (void)n_in; (void)out_size; (void)ws_size;
  const float* Dg = (const float*)d_in[0];
  const float* Sg = (const float*)d_in[1];
  const float* W[7];
  const float* Bb[7];
  for (int l = 0; l < 7; ++l) {
    W[l]  = (const float*)d_in[2 + 2 * l];
    Bb[l] = (const float*)d_in[3 + 2 * l];
  }
  _Float16* ws = (_Float16*)d_ws;
  float* out = (float*)d_out;

  prep_weights<<<dim3(16, 7, 1), 256, 0, stream>>>(W[0], W[1], W[2], W[3], W[4],
                                                   W[5], W[6], ws);
  aev_kernel<<<dim3(256, 1, 1), 128, SH_BYTES, stream>>>(
      Dg, Sg, ws, Bb[0], Bb[1], Bb[2], Bb[3], Bb[4], Bb[5], Bb[6], out);
}
